// Convolution_83459804496118
// MI455X (gfx1250) — compile-verified, matrix operand order checked by reading, not run on hardware
//
#include <hip/hip_runtime.h>
#include <hip/hip_bf16.h>

// ---- problem dims (fixed by the reference) ----
#define B_   8
#define CIN_ 576
#define C_   192
#define KK_  9
#define H_   56
#define W_   56
#define HW_  (H_ * W_)     // 3136
#define NOUT (C_ * KK_)    // 1728
#define MTOT (B_ * HW_)    // 25088 = 196 * 128

typedef _Float16 half4  __attribute__((ext_vector_type(4)));
typedef _Float16 half8  __attribute__((ext_vector_type(8)));
typedef _Float16 half16 __attribute__((ext_vector_type(16)));
typedef float    float8 __attribute__((ext_vector_type(8)));

// Load one 16x32 f16 WMMA operand fragment row from an LDS row laid out
// as contiguous halves in k. Per CDNA5 ISA layout tables: lanes 0-15 hold
// K={0..7,16..23}, lanes 16-31 hold K={8..15,24..31}; row index = lane&15.
static __device__ __forceinline__ half16 frag_row(const _Float16* row, int kb) {
    half8 lo = *(const half8*)(row + kb);        // k = kb .. kb+7
    half8 hi = *(const half8*)(row + kb + 16);   // k = kb+16 .. kb+23
    return __builtin_shufflevector(lo, hi, 0,1,2,3,4,5,6,7,8,9,10,11,12,13,14,15);
}

// ============================================================================
// Kernel 0a: transpose + convert  x (f32, [b][c][p])  ->  xh (f16, [b][p][c])
// so the GEMM's A-tiles are k-contiguous in global memory (pure b128 staging)
// and the whole A matrix is one contiguous [25088][576] array.
// ============================================================================
__global__ __launch_bounds__(256)
void cvt_x_transpose(const float* __restrict__ x, _Float16* __restrict__ xh) {
    __shared__ _Float16 t[32][33];
    const int p0 = blockIdx.x * 32, c0 = blockIdx.y * 32, b = blockIdx.z;
    const int tid = threadIdx.x;
    const float* xb = x + ((size_t)b * CIN_ + c0) * HW_ + p0;
    #pragma unroll
    for (int j = 0; j < 4; ++j) {
        int e = j * 256 + tid;
        int pp = e & 31, cc = e >> 5;            // coalesced in p
        t[cc][pp] = (_Float16)xb[(size_t)cc * HW_ + pp];
    }
    __syncthreads();
    _Float16* xo = xh + ((size_t)b * HW_ + p0) * CIN_ + c0;
    #pragma unroll
    for (int j = 0; j < 4; ++j) {
        int e = j * 256 + tid;
        int cc = e & 31, pp = e >> 5;            // coalesced in c
        xo[(size_t)pp * CIN_ + cc] = t[cc][pp];
    }
}

// Kernel 0b: fc_w (f32) -> fh (f16), same [o][k] layout. 995328 elems = 972*1024.
__global__ __launch_bounds__(256)
void cvt_fcw(const float* __restrict__ w, _Float16* __restrict__ wh) {
    int i = blockIdx.x * 256 + threadIdx.x;      // one float4 per thread
    float4 v = ((const float4*)w)[i];
    half4 h = {(_Float16)v.x, (_Float16)v.y, (_Float16)v.z, (_Float16)v.w};
    ((half4*)wh)[i] = h;
}

// ============================================================================
// Kernel 1: logits GEMM (f16 WMMA, f32 accumulate) + fused softmax over KK=9
//   attn[b,c,i,p] = softmax_i( sum_k fc_w[c*9+i, k] * x[b, k, p] )
// Block: 128 threads (4 waves). Tile: M=128 (flattened b*p), N=144.
// Each wave owns two 16-row M sub-tiles -> every B fragment feeds 2 WMMAs.
// Grid: (196 m-tiles, 12 n-groups). K = 576 = 9 chunks of 64.
// Epilogue (softmax + store) runs in two 64-row halves through a shared
// 64x144 f32 buffer unioned with the A/B staging tiles.
// ============================================================================
__global__ __launch_bounds__(128)
void dyconv_attn_gemm(const _Float16* __restrict__ xh,   // [25088][576]
                      const _Float16* __restrict__ fh,   // [1728][576]
                      float* __restrict__ attn) {
    __shared__ union SMem {
        struct { _Float16 A[128][64]; _Float16 Bt[144][64]; } st; // 34.0 KB
        float Ct[64][144];                                        // 36.9 KB
    } sm;

    const int mt  = blockIdx.x;             // 0..195
    const int ng  = blockIdx.y;             // 0..11
    const int r0  = mt * 128;               // global row base (b*HW + p)
    const int o0  = ng * 144;               // output-column base (= c0*9)
    const int c0  = ng * 16;                // channel base
    const int tid = threadIdx.x;
    const int wave = tid >> 5;
    const int lane = tid & 31;

    float8 acc[2][9] = {};                  // 2 M sub-tiles x 9 N-tiles per wave

    const int nrow = lane & 15;
    const int kb   = (lane < 16) ? 0 : 8;
    const int arow = wave * 16 + nrow;      // A row in sub-tile 0 (sub 1: +64)

    for (int ks = 0; ks < CIN_; ks += 64) {
        // --- stage A: 128x64 halves, 1024 x b128 chunks, 8 per thread ---
        #pragma unroll
        for (int j = 0; j < 8; ++j) {
            int q = j * 128 + tid;
            int r = q >> 3, c8 = (q & 7) * 8;
            *(half8*)&sm.st.A[r][c8] =
                *(const half8*)(xh + (size_t)(r0 + r) * CIN_ + ks + c8);
        }
        // --- stage B: 144x64 halves, 1152 x b128 chunks, 9 per thread ---
        #pragma unroll
        for (int j = 0; j < 9; ++j) {
            int q = j * 128 + tid;
            int r = q >> 3, c8 = (q & 7) * 8;
            *(half8*)&sm.st.Bt[r][c8] =
                *(const half8*)(fh + (size_t)(o0 + r) * CIN_ + ks + c8);
        }
        __syncthreads();

        #pragma unroll
        for (int kk = 0; kk < 64; kk += 32) {
            half16 a0 = frag_row(&sm.st.A[arow][kk], kb);
            half16 a1 = frag_row(&sm.st.A[64 + arow][kk], kb);
            #pragma unroll
            for (int t = 0; t < 9; ++t) {
                half16 bfr = frag_row(&sm.st.Bt[t * 16 + nrow][kk], kb);
                acc[0][t] = __builtin_amdgcn_wmma_f32_16x16x32_f16(
                    false, a0, false, bfr, (short)0, acc[0][t], false, false);
                acc[1][t] = __builtin_amdgcn_wmma_f32_16x16x32_f16(
                    false, a1, false, bfr, (short)0, acc[1][t], false, false);
            }
        }
        __syncthreads();
    }

    // ---- epilogue in two 64-row halves (Ct overlaps A/B staging) ----
    for (int h = 0; h < 2; ++h) {
        __syncthreads();   // h=0: redundant; h=1: fence prior-half Ct readers

        // spill accumulators (C/D layout: vgpr r -> M, lane -> N)
        #pragma unroll
        for (int t = 0; t < 9; ++t) {
            #pragma unroll
            for (int r = 0; r < 8; ++r) {
                int m = wave * 16 + ((lane >= 16) ? 8 : 0) + r;
                sm.Ct[m][t * 16 + nrow] = acc[h][t][r];
            }
        }
        __syncthreads();

        // softmax over groups of 9 columns, in place (1024 groups)
        #pragma unroll
        for (int j = 0; j < 8; ++j) {
            int g  = j * 128 + tid;          // 0..1023
            int m  = g & 63;
            int cl = g >> 6;                 // 0..15
            float* row = &sm.Ct[m][cl * 9];
            float mx = row[0];
            #pragma unroll
            for (int i = 1; i < 9; ++i) mx = fmaxf(mx, row[i]);
            float s = 0.f, e[9];
            #pragma unroll
            for (int i = 0; i < 9; ++i) { e[i] = __expf(row[i] - mx); s += e[i]; }
            float inv = 1.0f / s;
            #pragma unroll
            for (int i = 0; i < 9; ++i) row[i] = e[i] * inv;
        }
        __syncthreads();

        // coalesced store: row r = r0 + h*64 + m  ->  (b, p); attn[b, c, i, p]
        #pragma unroll
        for (int j = 0; j < 72; ++j) {
            int e  = j * 128 + tid;          // 0..9215
            int m  = e & 63;
            int pl = e >> 6;                 // 0..143 = cl*9 + i
            int cl = pl / 9, i = pl - cl * 9;
            int rg = r0 + h * 64 + m;
            int bb = rg / HW_;
            int p  = rg - bb * HW_;
            attn[(((size_t)bb * C_ + c0 + cl) * KK_ + i) * HW_ + p] = sm.Ct[m][pl];
        }
    }
}

// ============================================================================
// Kernel 2: fused per-channel pipeline. Output channel c uses only x channels
// 3c..3c+2, so both convs live in LDS; feat never touches HBM.
//   feat_i = relu(conv3x3(x[3c:3c+3], w1[i,c]))        (grouped, 3->1)
//   out   += conv3x3_dw(feat_i, w2[i,c]) * attn[b,c,i] (summed over i)
// Block: 256 threads, tile 28x28 (+halo2 on x, +halo1 on feat). f32-exact.
// Grid: (4 tiles, 192 channels, 8 batches).
// ============================================================================
__global__ __launch_bounds__(256)
void dyconv_fused_convs(const float* __restrict__ x,
                        const float* __restrict__ w1,
                        const float* __restrict__ w2,
                        const float* __restrict__ attn,
                        float* __restrict__ out) {
    __shared__ float xs[3][32][33];   // x tile + halo 2 (pad col vs bank conflicts)
    __shared__ float fs[30][31];      // feat tile + halo 1
    __shared__ float w1s[27];
    __shared__ float w2s[9];

    const int tile = blockIdx.x;                  // 0..3 (2x2 of 28)
    const int c    = blockIdx.y;
    const int b    = blockIdx.z;
    const int ty0  = (tile >> 1) * 28;
    const int tx0  = (tile & 1) * 28;
    const int tid  = threadIdx.x;

    // load x[b, 3c..3c+2] over [ty0-2, ty0+30) x [tx0-2, tx0+30), zero-padded
    const float* xb = x + ((size_t)b * CIN_ + 3 * c) * HW_;
    for (int e = tid; e < 3 * 32 * 32; e += 256) {
        int g  = e >> 10;
        int r  = e & 1023;
        int yy = r >> 5, xx = r & 31;
        int gy = ty0 - 2 + yy, gx = tx0 - 2 + xx;
        float v = 0.f;
        if (gy >= 0 && gy < H_ && gx >= 0 && gx < W_)
            v = xb[(size_t)g * HW_ + gy * W_ + gx];
        xs[g][yy][xx] = v;
    }

    float acc[4] = {0.f, 0.f, 0.f, 0.f};

    for (int i = 0; i < KK_; ++i) {
        __syncthreads();   // xs ready (i==0); prev-iter fs/w readers done (i>0)
        if (tid < 27)       w1s[tid]      = w1[((size_t)i * C_ + c) * 27 + tid];
        else if (tid < 36)  w2s[tid - 27] = w2[((size_t)i * C_ + c) * 9 + (tid - 27)];
        __syncthreads();

        // feat (30x30, origin ty0-1/tx0-1); positions outside image are ZERO
        // (reference zero-pads the conv2 input, not conv1 applied to padding).
        for (int e = tid; e < 900; e += 256) {
            int fy = e / 30, fx = e - fy * 30;
            int fgy = ty0 - 1 + fy, fgx = tx0 - 1 + fx;
            float v = 0.f;
            if (fgy >= 0 && fgy < H_ && fgx >= 0 && fgx < W_) {
                float s = 0.f;
                #pragma unroll
                for (int g = 0; g < 3; ++g)
                    #pragma unroll
                    for (int kh = 0; kh < 3; ++kh)
                        #pragma unroll
                        for (int kw = 0; kw < 3; ++kw)
                            s += xs[g][fy + kh][fx + kw] * w1s[g * 9 + kh * 3 + kw];
                v = s > 0.f ? s : 0.f;   // relu
            }
            fs[fy][fx] = v;
        }
        __syncthreads();

        // depthwise conv2 + softmax-weighted accumulation
        const float* ap = attn + (((size_t)b * C_ + c) * KK_ + i) * HW_;
        int ai = 0;
        for (int e = tid; e < 784; e += 256, ++ai) {
            int oy = e / 28, ox = e - oy * 28;
            float s = 0.f;
            #pragma unroll
            for (int kh = 0; kh < 3; ++kh)
                #pragma unroll
                for (int kw = 0; kw < 3; ++kw)
                    s += fs[oy + kh][ox + kw] * w2s[kh * 3 + kw];
            acc[ai] += s * ap[(ty0 + oy) * W_ + (tx0 + ox)];
        }
    }

    float* op = out + ((size_t)b * C_ + c) * HW_;
    int ai = 0;
    for (int e = tid; e < 784; e += 256, ++ai) {
        int oy = e / 28, ox = e - oy * 28;
        op[(ty0 + oy) * W_ + (tx0 + ox)] = acc[ai];
    }
}

// ============================================================================
extern "C" void kernel_launch(void* const* d_in, const int* in_sizes, int n_in,
                              void* d_out, int out_size, void* d_ws, size_t ws_size,
                              hipStream_t stream) {
    const float* x    = (const float*)d_in[0];   // (8, 576, 56, 56)
    const float* fc_w = (const float*)d_in[1];   // (1728, 576)
    const float* w1   = (const float*)d_in[2];   // (9, 192, 3, 3, 3)
    const float* w2   = (const float*)d_in[3];   // (9, 192, 3, 3)
    float* out  = (float*)d_out;                 // (8, 192, 56, 56)

    // workspace layout (all 16B-aligned):
    //   attn : 8*192*9*3136 f32  = 173,408,256 B
    //   xh   : 25088*576   f16   =  28,901,376 B   ([b][p][c] transposed)
    //   fh   : 1728*576    f16   =   1,990,656 B
    char* ws = (char*)d_ws;
    float*     attn = (float*)ws;
    _Float16*  xh   = (_Float16*)(ws + 173408256);
    _Float16*  fh   = (_Float16*)(ws + 173408256 + 28901376);

    // 0) precision/layout pre-passes
    dim3 g0(HW_ / 32, CIN_ / 32, B_);            // (98, 18, 8)
    cvt_x_transpose<<<g0, 256, 0, stream>>>(x, xh);
    cvt_fcw<<<(NOUT * CIN_) / 1024, 256, 0, stream>>>(fc_w, fh);

    // 1) WMMA GEMM + softmax -> attn   (M flattened across batches)
    dim3 g1(MTOT / 128, 12);                     // (196, 12)
    dyconv_attn_gemm<<<g1, 128, 0, stream>>>(xh, fh, attn);

    // 2) fused grouped conv -> relu -> depthwise conv -> attn-weighted sum
    dim3 g2(4, C_, B_);
    dyconv_fused_convs<<<g2, 256, 0, stream>>>(x, w1, w2, attn, out);
}
